// GE2E_Loss_43774306681159
// MI455X (gfx1250) — compile-verified, matrix operand order checked by reading, not run on hardware
//
#include <hip/hip_runtime.h>

#define N_SPK 2048
#define G_UTT 10
#define D_DIM 512
#define N_ROWS (N_SPK * G_UTT)      // 20480
#define EPS_COS   1e-8f
#define EPS_CLAMP 1e-6f
#define TILE_COLS 32                // columns per main-loop iteration (2 WMMA tiles)
#define BPAD 520                    // padded LDS row (520 halfs = 1040B) -> conflict-free ds reads
#define BUF_HALFS (TILE_COLS * BPAD)

typedef __attribute__((ext_vector_type(16))) __bf16       v16bf;
typedef __attribute__((ext_vector_type(8)))  float        v8f;
typedef __attribute__((ext_vector_type(4)))  unsigned int u32x4;
typedef __attribute__((ext_vector_type(8)))  int          i32x8;
typedef __attribute__((ext_vector_type(4)))  int          i32x4;

union BF16Frag { u32x4 u[2]; v16bf v; };

#if __has_builtin(__builtin_amdgcn_tensor_load_to_lds) && __has_builtin(__builtin_amdgcn_s_wait_tensorcnt)
#define GE2E_USE_TDM 1
#endif

static __device__ __forceinline__ unsigned short f2bf(float f) {
    unsigned int u = __float_as_uint(f);
    u += 0x7FFFu + ((u >> 16) & 1u);          // round-to-nearest-even
    return (unsigned short)(u >> 16);
}

#ifdef GE2E_USE_TDM
// Issue a TDM load of a TILE_COLS x D_DIM bf16 tile (row-major, stride D_DIM)
// into LDS at ldsByteOff, with TDM padding of 4 DWORDs every 256 DWORDs so the
// LDS image has BPAD(=520)-half rows (bank-conflict-free for per-lane reads).
static __device__ __forceinline__ void tdm_load_cols(const unsigned short* gsrc,
                                                     unsigned int ldsByteOff) {
    const unsigned long long ga = (unsigned long long)(uintptr_t)gsrc;
    u32x4 g0;
    g0.x = 1u;                                     // count=1, user mode, no gather
    g0.y = ldsByteOff;                             // lds_addr (bytes)
    g0.z = (unsigned int)ga;                       // global_addr[31:0]
    g0.w = (unsigned int)(ga >> 32) | (2u << 30);  // global_addr[56:32] | type=2 ("image")
    i32x8 g1;
    g1[0] = (1 << 16)      // data_size = 2 bytes
          | (1 << 20)      // pad_enable
          | (7 << 22)      // pad_interval: 256 DWORDs (= 512 halfs = one row)
          | (3 << 25);     // pad_amount: 4 DWORDs (= 8 halfs = BPAD-512)
    g1[1] = (int)(D_DIM << 16);   // tensor_dim0[15:0]  @ bits63:48
    g1[2] = (int)(N_SPK << 16);   // tensor_dim0[31:16]=0 | tensor_dim1[15:0]
    g1[3] = (int)(D_DIM << 16);   // tensor_dim1[31:16]=0 | tile_dim0
    g1[4] = TILE_COLS;            // tile_dim1 | tile_dim2=0
    g1[5] = D_DIM;                // tensor_dim0_stride[31:0]
    g1[6] = 0;                    // stride0 hi | tensor_dim1_stride lo (unused for 2D)
    g1[7] = 0;
    const i32x4 z4 = {0, 0, 0, 0};
#if __has_include(<hip/amd_detail/amd_gfx1250_TDM.h>)
    const i32x8 z8 = {0, 0, 0, 0, 0, 0, 0, 0};
    __builtin_amdgcn_tensor_load_to_lds(g0, g1, z4, z4, z8, 0);
#else
    __builtin_amdgcn_tensor_load_to_lds(g0, g1, z4, z4, 0);
#endif
}
#endif

// ---------------------------------------------------------------------------
// Kernel A: per speaker n -> sum_x[n,:], normalized centroid cn (bf16)
// ---------------------------------------------------------------------------
__global__ void ge2e_centroid_kernel(const float* __restrict__ x,
                                     float* __restrict__ sum_x,
                                     unsigned short* __restrict__ cn) {
    const int n   = blockIdx.x;
    const int tid = threadIdx.x;
    __shared__ float red[8];

    const int d0 = tid, d1 = tid + 256;
    const float* xb = x + (size_t)n * G_UTT * D_DIM;
    float s0 = 0.f, s1 = 0.f;
    #pragma unroll
    for (int g = 0; g < G_UTT; ++g) {
        s0 += xb[g * D_DIM + d0];
        s1 += xb[g * D_DIM + d1];
    }
    sum_x[(size_t)n * D_DIM + d0] = s0;
    sum_x[(size_t)n * D_DIM + d1] = s1;
    const float c0 = s0 * (1.0f / G_UTT);
    const float c1 = s1 * (1.0f / G_UTT);

    float acc = c0 * c0 + c1 * c1;
    #pragma unroll
    for (int m = 16; m; m >>= 1) acc += __shfl_xor(acc, m);
    if ((tid & 31) == 0) red[tid >> 5] = acc;
    __syncthreads();
    float norm2 = 0.f;
    #pragma unroll
    for (int i = 0; i < 8; ++i) norm2 += red[i];
    const float inv = 1.0f / fmaxf(sqrtf(norm2), EPS_COS);
    cn[(size_t)n * D_DIM + d0] = f2bf(c0 * inv);
    cn[(size_t)n * D_DIM + d1] = f2bf(c1 * inv);
}

// ---------------------------------------------------------------------------
// Kernel B: per row r=(n,g) -> exclusion cosine diag[r] (fp32), normalized xn (bf16)
// ---------------------------------------------------------------------------
__global__ void ge2e_excl_kernel(const float* __restrict__ x,
                                 const float* __restrict__ sum_x,
                                 unsigned short* __restrict__ xn,
                                 float* __restrict__ diag) {
    const int r   = blockIdx.x;
    const int n   = r / G_UTT;
    const int tid = threadIdx.x;
    __shared__ float redD[8], redX[8], redE[8];

    const int d0 = tid, d1 = tid + 256;
    const float* xr = x + (size_t)r * D_DIM;
    const float xv0 = xr[d0], xv1 = xr[d1];
    const float ev0 = (sum_x[(size_t)n * D_DIM + d0] - xv0) * (1.0f / (G_UTT - 1));
    const float ev1 = (sum_x[(size_t)n * D_DIM + d1] - xv1) * (1.0f / (G_UTT - 1));

    float pd = xv0 * ev0 + xv1 * ev1;
    float px = xv0 * xv0 + xv1 * xv1;
    float pe = ev0 * ev0 + ev1 * ev1;
    #pragma unroll
    for (int m = 16; m; m >>= 1) {
        pd += __shfl_xor(pd, m);
        px += __shfl_xor(px, m);
        pe += __shfl_xor(pe, m);
    }
    if ((tid & 31) == 0) { int w = tid >> 5; redD[w] = pd; redX[w] = px; redE[w] = pe; }
    __syncthreads();
    float td = 0.f, tx = 0.f, te = 0.f;
    #pragma unroll
    for (int i = 0; i < 8; ++i) { td += redD[i]; tx += redX[i]; te += redE[i]; }

    const float invx = 1.0f / fmaxf(sqrtf(tx), EPS_COS);
    if (tid == 0) diag[r] = td * invx / fmaxf(sqrtf(te), EPS_COS);
    xn[(size_t)r * D_DIM + d0] = f2bf(xv0 * invx);
    xn[(size_t)r * D_DIM + d1] = f2bf(xv1 * invx);
}

// ---------------------------------------------------------------------------
// Main kernel: fused WMMA GEMM (xn @ cn^T) + online softmax stats per row.
// Block = 8 wave32, each wave owns 16 rows (A resident in 128 VGPRs).
// 32-col B tile double-buffered in LDS; staged by the Tensor Data Mover
// (TENSORcnt) when available, overlapped with compute; one barrier/iter.
// ---------------------------------------------------------------------------
__global__ void __launch_bounds__(256, 1)
ge2e_sim_softmax_kernel(const unsigned short* __restrict__ xn,
                        const unsigned short* __restrict__ cn,
                        const float* __restrict__ diag,
                        const float* __restrict__ wp,
                        const float* __restrict__ bp,
                        float* __restrict__ partial) {
    extern __shared__ __align__(16) unsigned short blds[];   // 2 * BUF_HALFS halfs
    __shared__ float sred[16];

    const int tid     = threadIdx.x;
    const int wave    = tid >> 5;
    const int lane    = tid & 31;
    const int hi      = lane >> 4;          // half-group: K offset +8/+24
    const int l16     = lane & 15;          // A: row-in-tile, B/C: col-in-tile
    const int rowBase = blockIdx.x * 128 + wave * 16;
    const float w = *wp, b = *bp;

    // ---- resident A fragments: 16 K-chunks of 32 (ISA 16-bit A 16x32 layout)
    v16bf afrag[16];
    {
        const unsigned short* arow = xn + (size_t)(rowBase + l16) * D_DIM;
        #pragma unroll
        for (int kk = 0; kk < 16; ++kk) {
            const int k0 = kk * 32 + hi * 8;
            BF16Frag t;
            t.u[0] = *(const u32x4*)(arow + k0);        // K = k0 .. k0+7
            t.u[1] = *(const u32x4*)(arow + k0 + 16);   // K = k0+16 .. k0+23
            afrag[kk] = t.v;
        }
    }

    // ---- per-row online-softmax state (v -> row rowBase + v + 8*hi)
    float runMax[8], runSum[8], dval[8];
    int   bestIdx[8], labl[8];
    #pragma unroll
    for (int v = 0; v < 8; ++v) {
        const int r = rowBase + v + 8 * hi;
        runMax[v]  = -3.0e38f;
        runSum[v]  = 0.0f;
        bestIdx[v] = -1;
        dval[v]    = diag[r];
        labl[v]    = r / G_UTT;
    }

    const int NT = N_SPK / TILE_COLS;       // 64 iterations

#ifdef GE2E_USE_TDM
    if (wave == 0) tdm_load_cols(cn, (unsigned int)(uintptr_t)blds);
#else
    // register-staged double buffering: prefetch tile 0 into VGPRs
    u32x4 stage[8];
    #pragma unroll
    for (int i = 0; i < 8; ++i) {
        const int c = tid + i * 256;        // 32 cols * 64 chunks = 2048 chunks
        stage[i] = *(const u32x4*)(cn + (size_t)(c >> 6) * D_DIM + ((c & 63) << 3));
    }
#endif

    for (int t = 0; t < NT; ++t) {
        const int colBase = t * TILE_COLS;
        unsigned short* buf = blds + (t & 1) * BUF_HALFS;

#ifdef GE2E_USE_TDM
        __builtin_amdgcn_s_wait_tensorcnt(0);          // tile t landed (free for waves != 0)
        __syncthreads();                               // visible to all; prev buf reads done
        if (wave == 0 && t + 1 < NT)
            tdm_load_cols(cn + (size_t)(colBase + TILE_COLS) * D_DIM,
                          (unsigned int)(uintptr_t)(blds + ((t + 1) & 1) * BUF_HALFS));
#else
        #pragma unroll
        for (int i = 0; i < 8; ++i) {                  // commit staged tile t to LDS
            const int c = tid + i * 256;
            *(u32x4*)(buf + (c >> 6) * BPAD + ((c & 63) << 3)) = stage[i];
        }
        __syncthreads();
        if (t + 1 < NT) {                              // issue global loads for tile t+1
            const unsigned short* gnext = cn + (size_t)(colBase + TILE_COLS) * D_DIM;
            #pragma unroll
            for (int i = 0; i < 8; ++i) {
                const int c = tid + i * 256;
                stage[i] = *(const u32x4*)(gnext + (size_t)(c >> 6) * D_DIM + ((c & 63) << 3));
            }
        }
#endif

        // ---- full-K WMMA accumulation: two independent 16x16 tiles
        v8f acc0 = {}, acc1 = {};
        #pragma unroll
        for (int kk = 0; kk < 16; ++kk) {
            const int k0 = kk * 32 + hi * 8;
            const unsigned short* b0 = buf + l16 * BPAD + k0;
            const unsigned short* b1 = buf + (l16 + 16) * BPAD + k0;
            BF16Frag t0, t1;
            t0.u[0] = *(const u32x4*)(b0);
            t0.u[1] = *(const u32x4*)(b0 + 16);
            t1.u[0] = *(const u32x4*)(b1);
            t1.u[1] = *(const u32x4*)(b1 + 16);
            acc0 = __builtin_amdgcn_wmma_f32_16x16x32_bf16(
                       false, afrag[kk], false, t0.v, (short)0, acc0, false, false);
            acc1 = __builtin_amdgcn_wmma_f32_16x16x32_bf16(
                       false, afrag[kk], false, t1.v, (short)0, acc1, false, false);
        }

        // ---- post-process 32 columns: diag substitute, clamp, logits, online softmax
        #pragma unroll
        for (int v = 0; v < 8; ++v) {
            const int col0 = colBase + l16;
            const int col1 = col0 + 16;
            float s0 = acc0[v], s1 = acc1[v];
            if (col0 == labl[v]) s0 = dval[v];
            if (col1 == labl[v]) s1 = dval[v];
            const float l0 = fmaxf(s0, EPS_CLAMP) * w + b;
            const float l1 = fmaxf(s1, EPS_CLAMP) * w + b;

            float mval; int midx;                      // local max (tie -> lower index)
            if (l1 > l0) { mval = l1; midx = col1; } else { mval = l0; midx = col0; }
            #pragma unroll
            for (int m = 1; m < 16; m <<= 1) {
                const float ov = __shfl_xor(mval, m);
                const int   oi = __shfl_xor(midx, m);
                if (ov > mval || (ov == mval && oi < midx)) { mval = ov; midx = oi; }
            }
            const float newMax = fmaxf(runMax[v], mval);
            float e = __expf(l0 - newMax) + __expf(l1 - newMax);
            #pragma unroll
            for (int m = 1; m < 16; m <<= 1) e += __shfl_xor(e, m);
            runSum[v] = runSum[v] * __expf(runMax[v] - newMax) + e;
            if (mval > runMax[v]) bestIdx[v] = midx;   // strict > keeps first occurrence
            runMax[v] = newMax;
        }
    }

    // ---- finalize rows (state duplicated over each 16-lane group -> take l16==0)
    float lossAcc = 0.0f, corrAcc = 0.0f;
    if (l16 == 0) {
        #pragma unroll
        for (int v = 0; v < 8; ++v) {
            const float lse = runMax[v] + __logf(runSum[v]);
            const float lab = fmaxf(dval[v], EPS_CLAMP) * w + b;
            lossAcc += (lse - lab);
            corrAcc += (bestIdx[v] == labl[v]) ? 1.0f : 0.0f;
        }
    }
    #pragma unroll
    for (int m = 16; m; m >>= 1) {
        lossAcc += __shfl_xor(lossAcc, m);
        corrAcc += __shfl_xor(corrAcc, m);
    }
    if (lane == 0) { sred[wave * 2] = lossAcc; sred[wave * 2 + 1] = corrAcc; }
    __syncthreads();
    if (tid == 0) {
        float L = 0.f, C = 0.f;
        #pragma unroll
        for (int i = 0; i < 8; ++i) { L += sred[i * 2]; C += sred[i * 2 + 1]; }
        partial[blockIdx.x * 2]     = L;
        partial[blockIdx.x * 2 + 1] = C;
    }
}

// ---------------------------------------------------------------------------
// Deterministic final reduction of block partials -> (nloss, prec1)
// ---------------------------------------------------------------------------
__global__ void ge2e_finalize_kernel(const float* __restrict__ partial,
                                     float* __restrict__ out, int nblk) {
    if (threadIdx.x == 0 && blockIdx.x == 0) {
        float L = 0.f, C = 0.f;
        for (int i = 0; i < nblk; ++i) { L += partial[2 * i]; C += partial[2 * i + 1]; }
        out[0] = L / (float)N_ROWS;
        out[1] = C / (float)N_ROWS * 100.0f;
    }
}

extern "C" void kernel_launch(void* const* d_in, const int* in_sizes, int n_in,
                              void* d_out, int out_size, void* d_ws, size_t ws_size,
                              hipStream_t stream) {
    const float* x  = (const float*)d_in[0];
    const float* wp = (const float*)d_in[1];
    const float* bp = (const float*)d_in[2];
    float* out = (float*)d_out;

    char* ws = (char*)d_ws;
    const size_t off_sumx = 0;
    const size_t off_cn   = off_sumx + (size_t)N_SPK * D_DIM * sizeof(float);          // 4 MB
    const size_t off_xn   = off_cn   + (size_t)N_SPK * D_DIM * sizeof(unsigned short); // +2 MB
    const size_t off_diag = off_xn   + (size_t)N_ROWS * D_DIM * sizeof(unsigned short);// +20 MB
    const size_t off_part = off_diag + (size_t)N_ROWS * sizeof(float);                 // +80 KB

    float*          sum_x   = (float*)(ws + off_sumx);
    unsigned short* cn      = (unsigned short*)(ws + off_cn);
    unsigned short* xn      = (unsigned short*)(ws + off_xn);
    float*          diag    = (float*)(ws + off_diag);
    float*          partial = (float*)(ws + off_part);

    const int nblk = N_ROWS / 128;                      // 160
    const size_t dynLds = 2u * BUF_HALFS * sizeof(unsigned short);  // 66560 B

    ge2e_centroid_kernel<<<N_SPK, 256, 0, stream>>>(x, sum_x, cn);
    ge2e_excl_kernel<<<N_ROWS, 256, 0, stream>>>(x, sum_x, xn, diag);
    ge2e_sim_softmax_kernel<<<nblk, 256, dynLds, stream>>>(xn, cn, diag, wp, bp, partial);
    ge2e_finalize_kernel<<<1, 32, 0, stream>>>(partial, out, nblk);
}
